// MultiHeadAttention_87952340288009
// MI455X (gfx1250) — compile-verified, matrix operand order checked by reading, not run on hardware
//
#include <hip/hip_runtime.h>

// ---------------------------------------------------------------------------
// MHA forward for MI455X (gfx1250): bf16 WMMA (16x16x32) + f32 accumulation,
// flash-attention streaming softmax, all GEMMs on the matrix engine.
// Round 2: native bf16 converts (v_cvt_pk_bf16_f32 instead of integer RNE
// sequences), rcp instead of per-element divide, global_prefetch_b8 of the
// next tiles.
// ---------------------------------------------------------------------------

typedef __attribute__((ext_vector_type(16))) __bf16 bf16x16;
typedef __attribute__((ext_vector_type(8)))  float  floatx8;

union FragAB { unsigned int u[8]; bf16x16 v; };
union FragC  { float f[8]; floatx8 v; };

__device__ __forceinline__ unsigned short f2bf(float f) {
  __bf16 h = (__bf16)f;                      // native RNE convert on gfx1250
  return __builtin_bit_cast(unsigned short, h);
}
__device__ __forceinline__ unsigned int packbf2(float a, float b) {
  return (unsigned int)f2bf(a) | ((unsigned int)f2bf(b) << 16);
}
__device__ __forceinline__ floatx8 wmma_bf16(const FragAB& a, const FragAB& b,
                                             const floatx8& c) {
  return __builtin_amdgcn_wmma_f32_16x16x32_bf16(false, a.v, false, b.v,
                                                 (short)0, c, false, false);
}

// ===========================================================================
// Kernel 1: projection GEMM  C = X[4096,1024] @ W[1024,1024]  (f32 in)
// Output written as bf16 in [B=2, H=16, T=2048, Dk=64] layout.
// Block: 128 thr (4 waves), tile 64x64, wave tile 32x32, K-step 32.
// ===========================================================================
__global__ __launch_bounds__(128) void proj_gemm_kernel(
    const float* __restrict__ X, const float* __restrict__ W,
    unsigned short* __restrict__ Out) {
  __shared__ unsigned int ldsA[64 * 17];   // [m][k-pairs], row stride 17 dw
  __shared__ unsigned int ldsB[64 * 17];   // B^T: [n][k-pairs]
  const int tid  = threadIdx.x;
  const int lane = tid & 31, wave = tid >> 5;
  const int half = lane >> 4, l16 = lane & 15;
  const int wm = wave >> 1, wn = wave & 1;
  const int m0 = blockIdx.y * 64, n0 = blockIdx.x * 64;

  FragC c[2][2];
#pragma unroll
  for (int t = 0; t < 2; ++t)
#pragma unroll
    for (int t2 = 0; t2 < 2; ++t2)
#pragma unroll
      for (int r = 0; r < 8; ++r) c[t][t2].f[r] = 0.f;

  const int arr = tid >> 3, akc = (tid & 7) * 4;   // A staging map
  const int bkr = tid >> 4, bnc = (tid & 15) * 4;  // B staging map

  for (int kt = 0; kt < 32; ++kt) {
    const int kb = kt * 32;
    if (kt + 1 < 32) {   // prefetch next tiles (global_prefetch_b8)
      __builtin_prefetch(X + (size_t)(m0 + arr) * 1024 + kb + 32 + akc, 0, 1);
      __builtin_prefetch(W + (size_t)(kb + 32 + bkr) * 1024 + n0 + bnc, 0, 1);
    }
    {   // stage A tile (64x32 f32 -> bf16 pairs)
#pragma unroll
      for (int p = 0; p < 4; ++p) {
        const int row = p * 16 + arr;
        const float4 x = *reinterpret_cast<const float4*>(
            X + (size_t)(m0 + row) * 1024 + kb + akc);
        ldsA[row * 17 + (akc >> 1)]     = packbf2(x.x, x.y);
        ldsA[row * 17 + (akc >> 1) + 1] = packbf2(x.z, x.w);
      }
    }
    {   // stage B tile transposed: ldsB[n][k]
      unsigned short* B16 = reinterpret_cast<unsigned short*>(ldsB);
#pragma unroll
      for (int p = 0; p < 4; ++p) {
        const int k = p * 8 + bkr;
        const float4 w = *reinterpret_cast<const float4*>(
            W + (size_t)(kb + k) * 1024 + n0 + bnc);
        B16[(bnc + 0) * 34 + k] = f2bf(w.x);
        B16[(bnc + 1) * 34 + k] = f2bf(w.y);
        B16[(bnc + 2) * 34 + k] = f2bf(w.z);
        B16[(bnc + 3) * 34 + k] = f2bf(w.w);
      }
    }
    __syncthreads();
    FragAB a[2], b[2];
#pragma unroll
    for (int t = 0; t < 2; ++t) {
      const int base = (wm * 32 + t * 16 + l16) * 17 + half * 4;
#pragma unroll
      for (int v = 0; v < 4; ++v) {
        a[t].u[v]     = ldsA[base + v];
        a[t].u[v + 4] = ldsA[base + 8 + v];
      }
    }
#pragma unroll
    for (int t2 = 0; t2 < 2; ++t2) {
      const int base = (wn * 32 + t2 * 16 + l16) * 17 + half * 4;
#pragma unroll
      for (int v = 0; v < 4; ++v) {
        b[t2].u[v]     = ldsB[base + v];
        b[t2].u[v + 4] = ldsB[base + 8 + v];
      }
    }
#pragma unroll
    for (int t = 0; t < 2; ++t)
#pragma unroll
      for (int t2 = 0; t2 < 2; ++t2)
        c[t][t2].v = wmma_bf16(a[t], b[t2], c[t][t2].v);
    __syncthreads();
  }
  // epilogue: scatter into [B,H,T,64] bf16
#pragma unroll
  for (int t = 0; t < 2; ++t)
#pragma unroll
    for (int t2 = 0; t2 < 2; ++t2)
#pragma unroll
      for (int r = 0; r < 8; ++r) {
        const int m = m0 + wm * 32 + t * 16 + r + 8 * half;
        const int n = n0 + wn * 32 + t2 * 16 + l16;
        const int bb = m >> 11, tt = m & 2047, hh = n >> 6, dk = n & 63;
        Out[((size_t)(bb * 16 + hh) * 2048 + tt) * 64 + dk] =
            f2bf(c[t][t2].f[r]);
      }
}

// ===========================================================================
// Kernel 2: flash attention. grid = (T/64, B*H), block 128 (4 waves).
// Wave w owns q rows [qBase + 16w, +16). Streams 64-key tiles.
// ===========================================================================
__global__ __launch_bounds__(128) void attn_kernel(
    const unsigned short* __restrict__ Qb, const unsigned short* __restrict__ Kb,
    const unsigned short* __restrict__ Vb, unsigned short* __restrict__ Ab) {
  __shared__ unsigned int ldsK[64 * 33];        // [key][dk-pairs]
  __shared__ unsigned int ldsV[64 * 33];        // V^T: [dk][key-pairs]
  __shared__ unsigned int ldsP[4 * 16 * 33];    // per-wave P: [qrow][key-pairs]
  const int tid  = threadIdx.x;
  const int lane = tid & 31, wave = tid >> 5;
  const int half = lane >> 4, l16 = lane & 15;
  const int qt = blockIdx.x, bh = blockIdx.y;
  const int qBase = qt * 64;
  const size_t headOff = (size_t)bh * 2048 * 64;

  // preload Q A-fragments (16 rows x 64 dk = 2 k-steps), from global bf16
  FragAB qf[2];
  {
    const unsigned int* qrow = reinterpret_cast<const unsigned int*>(
        Qb + headOff + (size_t)(qBase + wave * 16 + l16) * 64);
#pragma unroll
    for (int s = 0; s < 2; ++s) {
      const int base = s * 16 + half * 4;
#pragma unroll
      for (int v = 0; v < 4; ++v) {
        qf[s].u[v]     = qrow[base + v];
        qf[s].u[v + 4] = qrow[base + 8 + v];
      }
    }
  }

  FragC acc[4];
#pragma unroll
  for (int j = 0; j < 4; ++j)
#pragma unroll
    for (int r = 0; r < 8; ++r) acc[j].f[r] = 0.f;
  float mrow[8], lrow[8];
#pragma unroll
  for (int r = 0; r < 8; ++r) { mrow[r] = -INFINITY; lrow[r] = 0.f; }

  const int srr = tid >> 3, sc8 = (tid & 7) * 8;   // K/V staging map
  const int nkt = qt + 1;       // causal: only key tiles <= q tile
  for (int kb_i = 0; kb_i < nkt; ++kb_i) {
    const int kBase = kb_i * 64;
    if (kb_i + 1 < nkt) {   // prefetch next K/V tile
      __builtin_prefetch(Kb + headOff + (size_t)(kBase + 64 + srr) * 64 + sc8, 0, 1);
      __builtin_prefetch(Vb + headOff + (size_t)(kBase + 64 + srr) * 64 + sc8, 0, 1);
    }
    {   // stage K tile row-major
#pragma unroll
      for (int p = 0; p < 4; ++p) {
        const int key = p * 16 + srr;
        const uint4 x = *reinterpret_cast<const uint4*>(
            Kb + headOff + (size_t)(kBase + key) * 64 + sc8);
        ldsK[key * 33 + (sc8 >> 1) + 0] = x.x;
        ldsK[key * 33 + (sc8 >> 1) + 1] = x.y;
        ldsK[key * 33 + (sc8 >> 1) + 2] = x.z;
        ldsK[key * 33 + (sc8 >> 1) + 3] = x.w;
      }
    }
    {   // stage V tile transposed: Vt[dk][key]
      unsigned short* V16 = reinterpret_cast<unsigned short*>(ldsV);
#pragma unroll
      for (int p = 0; p < 4; ++p) {
        const int key = p * 16 + srr;
        const uint4 x = *reinterpret_cast<const uint4*>(
            Vb + headOff + (size_t)(kBase + key) * 64 + sc8);
        const unsigned int w[4] = {x.x, x.y, x.z, x.w};
#pragma unroll
        for (int e = 0; e < 4; ++e) {
          V16[(sc8 + 2 * e + 0) * 66 + key] = (unsigned short)(w[e] & 0xFFFFu);
          V16[(sc8 + 2 * e + 1) * 66 + key] = (unsigned short)(w[e] >> 16);
        }
      }
    }
    __syncthreads();

    // S = Q K^T  (4 key n-tiles x 2 k-steps)
    FragC sc[4];
#pragma unroll
    for (int j = 0; j < 4; ++j) {
#pragma unroll
      for (int r = 0; r < 8; ++r) sc[j].f[r] = 0.f;
#pragma unroll
      for (int s = 0; s < 2; ++s) {
        FragAB kf;
        const int base = (16 * j + l16) * 33 + s * 16 + half * 4;
#pragma unroll
        for (int v = 0; v < 4; ++v) {
          kf.u[v]     = ldsK[base + v];
          kf.u[v + 4] = ldsK[base + 8 + v];
        }
        sc[j].v = wmma_bf16(qf[s], kf, sc[j].v);
      }
    }
    // scale 1/sqrt(64) + causal mask
#pragma unroll
    for (int j = 0; j < 4; ++j)
#pragma unroll
      for (int r = 0; r < 8; ++r) {
        const int kIdx = kBase + 16 * j + l16;
        const int qIdx = qBase + wave * 16 + r + 8 * half;
        const float s = sc[j].f[r] * 0.125f;
        sc[j].f[r] = (kIdx <= qIdx) ? s : -1e30f;
      }
    // online softmax (row = r + 8*half; reduce across 16-lane groups)
    float tmax[8];
#pragma unroll
    for (int r = 0; r < 8; ++r) {
      tmax[r] = fmaxf(fmaxf(sc[0].f[r], sc[1].f[r]),
                      fmaxf(sc[2].f[r], sc[3].f[r]));
#pragma unroll
      for (int off = 8; off >= 1; off >>= 1)
        tmax[r] = fmaxf(tmax[r], __shfl_xor(tmax[r], off, 32));
    }
    float al[8];
#pragma unroll
    for (int r = 0; r < 8; ++r) {
      const float mn = fmaxf(mrow[r], tmax[r]);
      al[r] = __expf(mrow[r] - mn);
      mrow[r] = mn;
    }
#pragma unroll
    for (int j = 0; j < 4; ++j)
#pragma unroll
      for (int r = 0; r < 8; ++r) sc[j].f[r] = __expf(sc[j].f[r] - mrow[r]);
    float ts[8];
#pragma unroll
    for (int r = 0; r < 8; ++r) {
      ts[r] = sc[0].f[r] + sc[1].f[r] + sc[2].f[r] + sc[3].f[r];
#pragma unroll
      for (int off = 8; off >= 1; off >>= 1)
        ts[r] += __shfl_xor(ts[r], off, 32);
      lrow[r] = lrow[r] * al[r] + ts[r];
    }
#pragma unroll
    for (int j = 0; j < 4; ++j)
#pragma unroll
      for (int r = 0; r < 8; ++r) acc[j].f[r] *= al[r];

    // P (C-layout) -> LDS, transpose into A-fragment layout
    {
      unsigned short* P16 = reinterpret_cast<unsigned short*>(ldsP);
#pragma unroll
      for (int j = 0; j < 4; ++j)
#pragma unroll
        for (int r = 0; r < 8; ++r)
          P16[(wave * 16 + r + 8 * half) * 66 + 16 * j + l16] =
              f2bf(sc[j].f[r]);
    }
    asm volatile("s_wait_dscnt 0" ::: "memory");   // intra-wave DS RAW

    FragAB pf[2];
#pragma unroll
    for (int s = 0; s < 2; ++s) {
      const int base = (wave * 16 + l16) * 33 + s * 16 + half * 4;
#pragma unroll
      for (int v = 0; v < 4; ++v) {
        pf[s].u[v]     = ldsP[base + v];
        pf[s].u[v + 4] = ldsP[base + 8 + v];
      }
    }
    // acc += P @ V   (4 dk n-tiles x 2 key k-steps)
#pragma unroll
    for (int j = 0; j < 4; ++j) {
#pragma unroll
      for (int s = 0; s < 2; ++s) {
        FragAB vf;
        const int base = (16 * j + l16) * 33 + s * 16 + half * 4;
#pragma unroll
        for (int v = 0; v < 4; ++v) {
          vf.u[v]     = ldsV[base + v];
          vf.u[v + 4] = ldsV[base + 8 + v];
        }
        acc[j].v = wmma_bf16(pf[s], vf, acc[j].v);
      }
    }
    __syncthreads();
  }

  // epilogue: A[b, t, h*64 + dk] bf16
  const int b = bh >> 4, h = bh & 15;
  float inv[8];
#pragma unroll
  for (int r = 0; r < 8; ++r) inv[r] = 1.0f / lrow[r];
#pragma unroll
  for (int j = 0; j < 4; ++j)
#pragma unroll
    for (int r = 0; r < 8; ++r) {
      const int qi = qBase + wave * 16 + r + 8 * half;
      Ab[(size_t)(b * 2048 + qi) * 1024 + h * 64 + 16 * j + l16] =
          f2bf(acc[j].f[r] * inv[r]);
    }
}

// ===========================================================================
// Kernel 3: output GEMM  out = A[4096,1024](bf16) @ Wo[1024,1024](f32) -> f32
// ===========================================================================
__global__ __launch_bounds__(128) void out_gemm_kernel(
    const unsigned short* __restrict__ A, const float* __restrict__ W,
    float* __restrict__ Out) {
  __shared__ unsigned int ldsA[64 * 17];
  __shared__ unsigned int ldsB[64 * 17];
  const int tid  = threadIdx.x;
  const int lane = tid & 31, wave = tid >> 5;
  const int half = lane >> 4, l16 = lane & 15;
  const int wm = wave >> 1, wn = wave & 1;
  const int m0 = blockIdx.y * 64, n0 = blockIdx.x * 64;

  FragC c[2][2];
#pragma unroll
  for (int t = 0; t < 2; ++t)
#pragma unroll
    for (int t2 = 0; t2 < 2; ++t2)
#pragma unroll
      for (int r = 0; r < 8; ++r) c[t][t2].f[r] = 0.f;

  const int arr = tid >> 2, acc8 = (tid & 3) * 8;  // A staging map
  const int bkr = tid >> 4, bnc = (tid & 15) * 4;  // B staging map

  for (int kt = 0; kt < 32; ++kt) {
    const int kb = kt * 32;
    if (kt + 1 < 32) {   // prefetch next tiles
      __builtin_prefetch(A + (size_t)(m0 + arr) * 1024 + kb + 32 + acc8, 0, 1);
      __builtin_prefetch(W + (size_t)(kb + 32 + bkr) * 1024 + n0 + bnc, 0, 1);
    }
    {   // stage A tile (already bf16)
#pragma unroll
      for (int p = 0; p < 2; ++p) {
        const int row = p * 32 + arr;
        const uint4 x = *reinterpret_cast<const uint4*>(
            A + (size_t)(m0 + row) * 1024 + kb + acc8);
        ldsA[row * 17 + (acc8 >> 1) + 0] = x.x;
        ldsA[row * 17 + (acc8 >> 1) + 1] = x.y;
        ldsA[row * 17 + (acc8 >> 1) + 2] = x.z;
        ldsA[row * 17 + (acc8 >> 1) + 3] = x.w;
      }
    }
    {   // stage Wo tile transposed bf16
      unsigned short* B16 = reinterpret_cast<unsigned short*>(ldsB);
#pragma unroll
      for (int p = 0; p < 4; ++p) {
        const int k = p * 8 + bkr;
        const float4 w = *reinterpret_cast<const float4*>(
            W + (size_t)(kb + k) * 1024 + n0 + bnc);
        B16[(bnc + 0) * 34 + k] = f2bf(w.x);
        B16[(bnc + 1) * 34 + k] = f2bf(w.y);
        B16[(bnc + 2) * 34 + k] = f2bf(w.z);
        B16[(bnc + 3) * 34 + k] = f2bf(w.w);
      }
    }
    __syncthreads();
    FragAB a[2], b[2];
#pragma unroll
    for (int t = 0; t < 2; ++t) {
      const int base = (wm * 32 + t * 16 + l16) * 17 + half * 4;
#pragma unroll
      for (int v = 0; v < 4; ++v) {
        a[t].u[v]     = ldsA[base + v];
        a[t].u[v + 4] = ldsA[base + 8 + v];
      }
    }
#pragma unroll
    for (int t2 = 0; t2 < 2; ++t2) {
      const int base = (wn * 32 + t2 * 16 + l16) * 17 + half * 4;
#pragma unroll
      for (int v = 0; v < 4; ++v) {
        b[t2].u[v]     = ldsB[base + v];
        b[t2].u[v + 4] = ldsB[base + 8 + v];
      }
    }
#pragma unroll
    for (int t = 0; t < 2; ++t)
#pragma unroll
      for (int t2 = 0; t2 < 2; ++t2)
        c[t][t2].v = wmma_bf16(a[t], b[t2], c[t][t2].v);
    __syncthreads();
  }
#pragma unroll
  for (int t = 0; t < 2; ++t)
#pragma unroll
    for (int t2 = 0; t2 < 2; ++t2)
#pragma unroll
      for (int r = 0; r < 8; ++r) {
        const int m = m0 + wm * 32 + t * 16 + r + 8 * half;
        const int n = n0 + wn * 32 + t2 * 16 + l16;
        Out[(size_t)m * 1024 + n] = c[t][t2].f[r];
      }
}

// ===========================================================================
extern "C" void kernel_launch(void* const* d_in, const int* in_sizes, int n_in,
                              void* d_out, int out_size, void* d_ws,
                              size_t ws_size, hipStream_t stream) {
  (void)in_sizes; (void)n_in; (void)out_size; (void)ws_size;
  const float* X  = (const float*)d_in[0];
  const float* Wq = (const float*)d_in[1];
  const float* Wk = (const float*)d_in[2];
  const float* Wv = (const float*)d_in[3];
  const float* Wo = (const float*)d_in[4];
  float* out = (float*)d_out;

  // workspace: Q,K,V in [B,H,T,64] bf16 (8 MiB each) + A in [B*T,1024] bf16
  const size_t HEAD_ELEMS = (size_t)2 * 16 * 2048 * 64;   // 4,194,304
  unsigned short* Qb = (unsigned short*)d_ws;
  unsigned short* Kb = Qb + HEAD_ELEMS;
  unsigned short* Vb = Kb + HEAD_ELEMS;
  unsigned short* Ab = Vb + HEAD_ELEMS;

  const dim3 gGemm(16, 64), blk(128);
  proj_gemm_kernel<<<gGemm, blk, 0, stream>>>(X, Wq, Qb);
  proj_gemm_kernel<<<gGemm, blk, 0, stream>>>(X, Wk, Kb);
  proj_gemm_kernel<<<gGemm, blk, 0, stream>>>(X, Wv, Vb);
  attn_kernel<<<dim3(32, 32), blk, 0, stream>>>(Qb, Kb, Vb, Ab);
  out_gemm_kernel<<<gGemm, blk, 0, stream>>>(Ab, Wo, out);
}
